// EGNNLayer_84138409328671
// MI455X (gfx1250) — compile-verified, matrix operand order checked by reading, not run on hardware
//
#include <hip/hip_runtime.h>
#include <math.h>

#define BB 2
#define NN 384
#define ND 128
#define ED 32
#define HD 128
#define NCHUNK 12      // 384 j's / (4 waves * 8 j per wave)
#define JW 8           // j's handled per wave
#define WAVES 4
#define ITILE 16
#define KS1 9          // K 0..287 via WMMA; K=288 (dist^2) folded in via VALU
#define KS2 4          // 128/32

typedef __attribute__((ext_vector_type(16))) __bf16 v16bf;
typedef __attribute__((ext_vector_type(8)))  float  v8f;

union BF16x16 { unsigned short s[16]; uint4 u[2]; v16bf v; };

__device__ __forceinline__ unsigned short f2bf(float f) {
    union { float f; unsigned u; } c; c.f = f;
    unsigned u = c.u;
    u += 0x7FFFu + ((u >> 16) & 1u);        // round-to-nearest-even
    return (unsigned short)(u >> 16);
}
__device__ __forceinline__ float silu_f(float x) { return x / (1.f + __expf(-x)); }

// K-within-step position held by (lane, element e) for the 16x32 bf16 A/B fragment:
//   lane<16: e<8 -> K=e,      e>=8 -> K=e+8   (K 0..7, 16..23)
//   lane>=16:e<8 -> K=e+8,    e>=8 -> K=e+16  (K 8..15, 24..31)
__device__ __forceinline__ int klocal(int lane, int e) {
    return ((lane & 16) ? 8 : 0) + ((e < 8) ? e : (e + 8));
}

// 8 WMMAs across the 128-wide output for one K-step; B frags streamed from
// the pre-swizzled global fragment buffer (2 x b128 per fragment per lane).
__device__ __forceinline__ void wmma_row(v8f* acc, const BF16x16& A,
                                         const uint4* wf4, int ks, int lane) {
#pragma unroll
    for (int nt = 0; nt < 8; ++nt) {
        BF16x16 Bf;
        int bi = ((ks * 8 + nt) * 32 + lane) * 2;
        Bf.u[0] = wf4[bi]; Bf.u[1] = wf4[bi + 1];
        acc[nt] = __builtin_amdgcn_wmma_f32_16x16x32_bf16(
            false, A.v, false, Bf.v, (short)0, acc[nt], false, false);
    }
}

// ---------------------------------------------------------------------------
// Prep: f32 weights -> bf16 fragments pre-swizzled into WMMA B layout.
// Storage: frag[kstep][ntile][lane][e], each lane's 16 bf16 contiguous (32B).
// ---------------------------------------------------------------------------
__global__ void prep_weights(const float* __restrict__ w1,
                             const float* __restrict__ w2,
                             const float* __restrict__ w3,
                             unsigned short* __restrict__ w1f,
                             unsigned short* __restrict__ w2f,
                             unsigned short* __restrict__ w3f) {
    int idx = blockIdx.x * 256 + threadIdx.x;
    const int n1 = KS1 * 8 * 32 * 16;   // 36864  (K 0..287)
    const int n2 = KS2 * 8 * 32 * 16;   // 16384
    if (idx < n1) {
        int p = idx;
        int e = p & 15, lane = (p >> 4) & 31, nt = (p >> 9) & 7, ks = p >> 12;
        int K = ks * 32 + klocal(lane, e);
        int n = nt * 16 + (lane & 15);
        w1f[p] = f2bf(w1[K * HD + n]);
    } else if (idx < n1 + n2) {
        int p = idx - n1;
        int e = p & 15, lane = (p >> 4) & 31, nt = (p >> 9) & 7, ks = p >> 12;
        int K = ks * 32 + klocal(lane, e);
        int n = nt * 16 + (lane & 15);
        w2f[p] = f2bf(w2[K * HD + n]);
    } else if (idx < n1 + 2 * n2) {
        int p = idx - n1 - n2;
        int e = p & 15, lane = (p >> 4) & 31, nt = (p >> 9) & 7, ks = p >> 12;
        int K = ks * 32 + klocal(lane, e);
        int n = nt * 16 + (lane & 15);
        w3f[p] = f2bf(w3[K * HD + n]);
    }
}

// ---------------------------------------------------------------------------
// Edge kernel: fused msg-MLP (2 layers) + coord-MLP per edge tile via WMMA.
// Block = 128 threads (4 waves). Each wave handles JW j's; each M-tile is
// 16 i-rows at one fixed j. h_agg/x_agg accumulate in per-wave LDS buffers
// (distinct address per lane -> deterministic), reduced in fixed order.
// ---------------------------------------------------------------------------
__global__ void __launch_bounds__(128)
egnn_edge_kernel(const float* __restrict__ h,
                 const float* __restrict__ xg,
                 const float* __restrict__ ea,
                 const float* __restrict__ mw1,   // for W1[288,:] (dist row)
                 const float* __restrict__ mb1,
                 const float* __restrict__ mb2,
                 const float* __restrict__ cb1,
                 const float* __restrict__ cw2,
                 const int*   __restrict__ adj,
                 const unsigned short* __restrict__ w1f,
                 const unsigned short* __restrict__ w2f,
                 const unsigned short* __restrict__ w3f,
                 float* __restrict__ ph,      // [B*N][NCHUNK][HD]
                 float* __restrict__ px)      // [B*N][NCHUNK][3]
{
    __shared__ __align__(16) unsigned short hi_bf[ITILE * HD];          // 4 KB
    __shared__ __align__(16) unsigned short hj_bf[WAVES * HD];          // 1 KB
    __shared__ __align__(16) unsigned short mstage[WAVES * ITILE * HD]; // 16 KB
    __shared__ float haccL[WAVES * ITILE * HD];                         // 32 KB
    __shared__ float xaccL[WAVES * ITILE * 3];
    __shared__ float xi[ITILE * 3];
    __shared__ float dsqb[WAVES * ITILE];
    __shared__ float adjb[WAVES * ITILE];

    const int t    = threadIdx.x;
    const int lane = t & 31;
    const int wave = t >> 5;
    const int row  = lane & 15;
    const int half = (lane >> 4) & 1;
    const int b    = blockIdx.z;
    const int i0   = blockIdx.y * ITILE;
    const int chunk = blockIdx.x;

    // stage h_i tile (bf16), x_i; zero LDS accumulators
    for (int idx = t; idx < ITILE * HD; idx += 128) {
        int r = idx >> 7, c = idx & 127;
        hi_bf[idx] = f2bf(h[(b * NN + i0 + r) * HD + c]);
    }
    for (int idx = t; idx < WAVES * ITILE * HD; idx += 128) haccL[idx] = 0.f;
    for (int idx = t; idx < WAVES * ITILE * 3; idx += 128)  xaccL[idx] = 0.f;
    if (t < ITILE * 3) xi[t] = xg[(b * NN + i0 + (t / 3)) * 3 + (t % 3)];
    __syncthreads();

    float bias1[8], bias2[8], bias3[8], cw4[8], w288[8];
#pragma unroll
    for (int nt = 0; nt < 8; ++nt) {
        bias1[nt] = mb1[nt * 16 + row];
        bias2[nt] = mb2[nt * 16 + row];
        bias3[nt] = cb1[nt * 16 + row];
        cw4[nt]   = cw2[nt * 16 + row];
        w288[nt]  = mw1[288 * HD + nt * 16 + row];  // dist^2 rank-1 row
    }

    const uint4* w1f4 = (const uint4*)w1f;
    const uint4* w2f4 = (const uint4*)w2f;
    const uint4* w3f4 = (const uint4*)w3f;

#pragma unroll 1
    for (int r = 0; r < JW; ++r) {
        const int j = chunk * (WAVES * JW) + wave * JW + r;

        const float xj0 = xg[(b * NN + j) * 3 + 0];
        const float xj1 = xg[(b * NN + j) * 3 + 1];
        const float xj2 = xg[(b * NN + j) * 3 + 2];

        if (lane < 16) {
            const float4* hp = (const float4*)(h + (b * NN + j) * HD);
            float4 a0 = hp[row * 2], a1 = hp[row * 2 + 1];
            int base = wave * HD + row * 8;
            hj_bf[base + 0] = f2bf(a0.x); hj_bf[base + 1] = f2bf(a0.y);
            hj_bf[base + 2] = f2bf(a0.z); hj_bf[base + 3] = f2bf(a0.w);
            hj_bf[base + 4] = f2bf(a1.x); hj_bf[base + 5] = f2bf(a1.y);
            hj_bf[base + 6] = f2bf(a1.z); hj_bf[base + 7] = f2bf(a1.w);
            float d0 = xi[row * 3 + 0] - xj0;
            float d1 = xi[row * 3 + 1] - xj1;
            float d2 = xi[row * 3 + 2] - xj2;
            dsqb[wave * 16 + row] = d0 * d0 + d1 * d1 + d2 * d2;
            adjb[wave * 16 + row] = (float)adj[(b * NN + i0 + row) * NN + j];
        }
        __syncthreads();

        // ---- layer 1: msg_in[16x289] @ W1 -> [16x128]
        v8f acc[8];
#pragma unroll
        for (int nt = 0; nt < 8; ++nt)
#pragma unroll
            for (int v = 0; v < 8; ++v) acc[nt][v] = 0.f;

#pragma unroll 1
        for (int ks = 0; ks < 4; ++ks) {            // h_i features 0..127
            BF16x16 A;
            int e0 = row * HD + ks * 32 + half * 8;
            A.u[0] = *(const uint4*)(hi_bf + e0);
            A.u[1] = *(const uint4*)(hi_bf + e0 + 16);
            wmma_row(acc, A, w1f4, ks, lane);
        }
#pragma unroll 1
        for (int ks = 0; ks < 4; ++ks) {            // h_j features 128..255
            BF16x16 A;
            int e0 = wave * HD + ks * 32 + half * 8;
            A.u[0] = *(const uint4*)(hj_bf + e0);
            A.u[1] = *(const uint4*)(hj_bf + e0 + 16);
            wmma_row(acc, A, w1f4, ks + 4, lane);
        }
        {                                           // edge_attr 256..287
            BF16x16 A;
            const float4* ep =
                (const float4*)(ea + ((b * NN + i0 + row) * NN + j) * ED);
            float4 ra0 = ep[half * 2],     ra1 = ep[half * 2 + 1];
            float4 rb0 = ep[4 + half * 2], rb1 = ep[5 + half * 2];
            A.s[0] = f2bf(ra0.x); A.s[1] = f2bf(ra0.y);
            A.s[2] = f2bf(ra0.z); A.s[3] = f2bf(ra0.w);
            A.s[4] = f2bf(ra1.x); A.s[5] = f2bf(ra1.y);
            A.s[6] = f2bf(ra1.z); A.s[7] = f2bf(ra1.w);
            A.s[8]  = f2bf(rb0.x); A.s[9]  = f2bf(rb0.y);
            A.s[10] = f2bf(rb0.z); A.s[11] = f2bf(rb0.w);
            A.s[12] = f2bf(rb1.x); A.s[13] = f2bf(rb1.y);
            A.s[14] = f2bf(rb1.z); A.s[15] = f2bf(rb1.w);
            wmma_row(acc, A, w1f4, 8, lane);
        }

        // epilogue 1: + bias + dist^2 rank-1 term, silu, stage bf16
        float dsl[8];
#pragma unroll
        for (int v = 0; v < 8; ++v) dsl[v] = dsqb[wave * 16 + v + half * 8];
#pragma unroll
        for (int nt = 0; nt < 8; ++nt)
#pragma unroll
            for (int v = 0; v < 8; ++v) {
                float val = silu_f(acc[nt][v] + bias1[nt] + dsl[v] * w288[nt]);
                mstage[wave * ITILE * HD + (v + half * 8) * HD + nt * 16 + row] =
                    f2bf(val);
            }
        __syncthreads();

        // ---- layer 2: m1 @ W2
#pragma unroll
        for (int nt = 0; nt < 8; ++nt)
#pragma unroll
            for (int v = 0; v < 8; ++v) acc[nt][v] = 0.f;
#pragma unroll 1
        for (int ks = 0; ks < KS2; ++ks) {
            BF16x16 A;
            int e0 = wave * ITILE * HD + row * HD + ks * 32 + half * 8;
            A.u[0] = *(const uint4*)(mstage + e0);
            A.u[1] = *(const uint4*)(mstage + e0 + 16);
            wmma_row(acc, A, w2f4, ks, lane);
        }

        float adjm[8];
#pragma unroll
        for (int v = 0; v < 8; ++v) adjm[v] = adjb[wave * 16 + v + half * 8];
        __syncthreads();   // all reads of m1 done before overwriting mstage

        // m = silu(.+b2)*adj ; accumulate h_agg into LDS ; restage for coord
#pragma unroll
        for (int nt = 0; nt < 8; ++nt)
#pragma unroll
            for (int v = 0; v < 8; ++v) {
                float val = silu_f(acc[nt][v] + bias2[nt]) * adjm[v];
                int off = (v + half * 8) * HD + nt * 16 + row;
                atomicAdd(&haccL[wave * ITILE * HD + off], val);
                mstage[wave * ITILE * HD + off] = f2bf(val);
            }
        __syncthreads();

        // ---- layer 3: m @ coord_w1, then dot with coord_w2
#pragma unroll
        for (int nt = 0; nt < 8; ++nt)
#pragma unroll
            for (int v = 0; v < 8; ++v) acc[nt][v] = 0.f;
#pragma unroll 1
        for (int ks = 0; ks < KS2; ++ks) {
            BF16x16 A;
            int e0 = wave * ITILE * HD + row * HD + ks * 32 + half * 8;
            A.u[0] = *(const uint4*)(mstage + e0);
            A.u[1] = *(const uint4*)(mstage + e0 + 16);
            wmma_row(acc, A, w3f4, ks, lane);
        }
        float cwp[8];
#pragma unroll
        for (int v = 0; v < 8; ++v) cwp[v] = 0.f;
#pragma unroll
        for (int nt = 0; nt < 8; ++nt)
#pragma unroll
            for (int v = 0; v < 8; ++v)
                cwp[v] += silu_f(acc[nt][v] + bias3[nt]) * cw4[nt];

#pragma unroll
        for (int v = 0; v < 8; ++v) {
            float s = cwp[v];
            s += __shfl_xor(s, 1, 16);
            s += __shfl_xor(s, 2, 16);
            s += __shfl_xor(s, 4, 16);
            s += __shfl_xor(s, 8, 16);
            int M = v + half * 8;
            float dsq = dsqb[wave * 16 + M];
            float cwv = s / (sqrtf(dsq + 1e-8f) + 1.f) * adjm[v];
            if (row < 3) {
                float xjc = (row == 0) ? xj0 : ((row == 1) ? xj1 : xj2);
                atomicAdd(&xaccL[wave * ITILE * 3 + M * 3 + row],
                          cwv * (xi[M * 3 + row] - xjc));
            }
        }
    }

    // fixed-order 4-way reduction of per-wave LDS accumulators
    __syncthreads();
    for (int idx = t; idx < ITILE * HD; idx += 128) {
        float s = ((haccL[idx] + haccL[ITILE * HD + idx]) +
                   haccL[2 * ITILE * HD + idx]) + haccL[3 * ITILE * HD + idx];
        int M = idx >> 7, n = idx & 127;
        ph[((b * NN + i0 + M) * NCHUNK + chunk) * HD + n] = s;
    }
    if (t < ITILE * 3) {
        float s = ((xaccL[t] + xaccL[48 + t]) + xaccL[96 + t]) + xaccL[144 + t];
        int M = t / 3, c = t % 3;
        px[((b * NN + i0 + M) * NCHUNK + chunk) * 3 + c] = s;
    }
}

// ---------------------------------------------------------------------------
// Node kernel: reduce partials (fixed order), node MLP, LayerNorm, x update.
// ---------------------------------------------------------------------------
__global__ void __launch_bounds__(128)
egnn_node_kernel(const float* __restrict__ h, const float* __restrict__ xg,
                 const float* __restrict__ amask,
                 const float* __restrict__ w1, const float* __restrict__ b1,
                 const float* __restrict__ w2, const float* __restrict__ b2,
                 const float* __restrict__ gamma, const float* __restrict__ beta,
                 const float* __restrict__ ph, const float* __restrict__ px,
                 float* __restrict__ out_h, float* __restrict__ out_x)
{
    __shared__ float cat[ND + HD];
    __shared__ float us[HD];
    __shared__ float red[HD];
    const int rowi = blockIdx.x;
    const int t = threadIdx.x;

    float hval = h[rowi * ND + t];
    cat[t] = hval;
    float s = 0.f;
    for (int c = 0; c < NCHUNK; ++c) s += ph[(rowi * NCHUNK + c) * HD + t];
    cat[ND + t] = s;
    __syncthreads();

    float u = b1[t];
    for (int k = 0; k < ND + HD; ++k) u += cat[k] * w1[k * HD + t];
    u = silu_f(u);
    us[t] = u;
    __syncthreads();

    float hu = b2[t];
    for (int k = 0; k < HD; ++k) hu += us[k] * w2[k * ND + t];
    float y = hval + hu;

    red[t] = y; __syncthreads();
    for (int sft = 64; sft > 0; sft >>= 1) {
        if (t < sft) red[t] += red[t + sft];
        __syncthreads();
    }
    float mu = red[0] * (1.f / ND);
    __syncthreads();
    float d = y - mu;
    red[t] = d * d; __syncthreads();
    for (int sft = 64; sft > 0; sft >>= 1) {
        if (t < sft) red[t] += red[t + sft];
        __syncthreads();
    }
    float var = red[0] * (1.f / ND);
    out_h[rowi * ND + t] = d * rsqrtf(var + 1e-3f) * gamma[t] + beta[t];

    if (t < 3) {
        float xs = 0.f;
        for (int c = 0; c < NCHUNK; ++c) xs += px[(rowi * NCHUNK + c) * 3 + t];
        out_x[rowi * 3 + t] = xg[rowi * 3 + t] + xs * amask[rowi];
    }
}

extern "C" void kernel_launch(void* const* d_in, const int* in_sizes, int n_in,
                              void* d_out, int out_size, void* d_ws, size_t ws_size,
                              hipStream_t stream) {
    (void)in_sizes; (void)n_in; (void)out_size; (void)ws_size;
    const float* h   = (const float*)d_in[0];
    const float* x   = (const float*)d_in[1];
    const float* ea  = (const float*)d_in[2];
    const float* am  = (const float*)d_in[3];
    const float* mw1 = (const float*)d_in[4];
    const float* mb1 = (const float*)d_in[5];
    const float* mw2 = (const float*)d_in[6];
    const float* mb2 = (const float*)d_in[7];
    const float* nw1 = (const float*)d_in[8];
    const float* nb1 = (const float*)d_in[9];
    const float* nw2 = (const float*)d_in[10];
    const float* nb2 = (const float*)d_in[11];
    const float* cw1 = (const float*)d_in[12];
    const float* cb1 = (const float*)d_in[13];
    const float* cw2 = (const float*)d_in[14];
    const float* gam = (const float*)d_in[15];
    const float* bet = (const float*)d_in[16];
    const int*   adj = (const int*)d_in[17];

    char* ws = (char*)d_ws;
    unsigned short* w1f = (unsigned short*)ws;                        // 73728 B
    unsigned short* w2f = (unsigned short*)(ws + 73728);              // 32768 B
    unsigned short* w3f = (unsigned short*)(ws + 73728 + 32768);      // 32768 B
    float* ph = (float*)(ws + 139264);                                // 4.72 MB
    float* px = (float*)(ws + 139264 + (size_t)BB * NN * NCHUNK * HD * 4);

    prep_weights<<<272, 256, 0, stream>>>(mw1, mw2, cw1, w1f, w2f, w3f);

    dim3 grid(NCHUNK, NN / ITILE, BB);
    egnn_edge_kernel<<<grid, 128, 0, stream>>>(h, x, ea, mw1, mb1, mb2, cb1, cw2,
                                               adj, w1f, w2f, w3f, ph, px);

    float* out_h = (float*)d_out;
    float* out_x = out_h + BB * NN * ND;
    egnn_node_kernel<<<BB * NN, 128, 0, stream>>>(h, x, am, nw1, nb1, nw2, nb2,
                                                  gam, bet, ph, px, out_h, out_x);
}
